// MultiHeadAttention_59554016526735
// MI455X (gfx1250) — matrix operand order repaired and checked by reading, not compiled, not run on hardware
//
#include <hip/hip_runtime.h>
#include <hip/hip_bf16.h>

// MI455X / gfx1250, wave32. bf16 WMMA (V_WMMA_F32_16X16X32_BF16) with fp32 accum.
// Pipeline:
//   [P] one-time: X -> bf16 (row-major), W -> bf16 transposed (B-operand native)
//   [1] QKV projection -> bf16 q (pre-scaled), k (row-major), v^T ([d][s])
//   [2] flash attention (S^T trick, register-resident softmax) -> bf16 o
//   [3] output projection -> fp32 d_out
// All WMMA operands are contiguous 16B loads; loads of each K-step are hoisted
// ahead of the WMMA group so the scheduler can overlap them (partial waits).

typedef __attribute__((ext_vector_type(16))) __bf16 bf16x16;
typedef __attribute__((ext_vector_type(8)))  __bf16 bf16x8;
typedef __attribute__((ext_vector_type(8)))  float  f32x8;

#define E_DIM 768
#define H_NUM 12
#define D_DIM 64
#define S_LEN 2048
#define B_NUM 2
#define PLANE ((size_t)S_LEN * D_DIM)               // one (b,h) plane (elements)
#define X_SZ  ((size_t)B_NUM * S_LEN * E_DIM)       // one activation tensor
#define W_SZ  ((size_t)H_NUM * E_DIM * D_DIM)       // one per-head weight set
#define WO_SZ ((size_t)E_DIM * E_DIM)
#define P_SZ  ((size_t)B_NUM * H_NUM * PLANE)       // one of q/k/v/o

static __device__ inline f32x8 wmma_bf16f32(bf16x16 a, bf16x16 b, f32x8 c) {
  // (neg_a, A, neg_b, B, c_mod, C, reuse_a, reuse_b)
  return __builtin_amdgcn_wmma_f32_16x16x32_bf16(false, a, false, b, (short)0, c, false, false);
}

static __device__ inline f32x8 fzero8() {
  f32x8 z = {0.f, 0.f, 0.f, 0.f, 0.f, 0.f, 0.f, 0.f};
  return z;
}

// A-fragment (16x32) from row-major bf16; caller pre-offsets
// p = base + (lane&15)*ld + kbase + ((lane&16)>>1).
// A layout: element i holds k = ((i>>3)<<4) + (i&7); runs at +0 and +16.
static __device__ inline bf16x16 load_a_bf16(const __bf16* __restrict__ p) {
  bf16x8 lo = *(const bf16x8*)(p);
  bf16x8 hi = *(const bf16x8*)(p + 16);
  return __builtin_shufflevector(lo, hi, 0, 1, 2, 3, 4, 5, 6, 7,
                                 8, 9, 10, 11, 12, 13, 14, 15);
}

// B-fragment (32x16) when a lane's 16 K values are contiguous (transposed
// weights / q^T / v^T). p = base + n*ld + kbase + ((lane&16)>>1). Element i: k = ((i>>3)<<4) + (i&7).
static __device__ inline bf16x16 load_b_bf16(const __bf16* __restrict__ p) {
  bf16x8 lo = *(const bf16x8*)(p);
  bf16x8 hi = *(const bf16x8*)(p + 16);
  return __builtin_shufflevector(lo, hi, 0, 1, 2, 3, 4, 5, 6, 7,
                                 8, 9, 10, 11, 12, 13, 14, 15);
}

// ---------------------------------------------------------------------------
// Prep: fp32 -> bf16 elementwise (n8 = n/8 vector chunks).
__global__ __launch_bounds__(256) void mha_cvt_bf16(
    const float* __restrict__ src, __bf16* __restrict__ dst, int n8) {
  int i = blockIdx.x * 256 + threadIdx.x;
  if (i >= n8) return;
  const float4* s = (const float4*)src + (size_t)i * 2;
  float4 f0 = s[0], f1 = s[1];
  bf16x8 r;
  r[0] = (__bf16)f0.x; r[1] = (__bf16)f0.y; r[2] = (__bf16)f0.z; r[3] = (__bf16)f0.w;
  r[4] = (__bf16)f1.x; r[5] = (__bf16)f1.y; r[6] = (__bf16)f1.z; r[7] = (__bf16)f1.w;
  *((bf16x8*)dst + i) = r;
}

// Prep: batched transpose+convert  W[batch][R][C] f32 -> Wt[batch][C][R] bf16.
__global__ __launch_bounds__(256) void mha_transpose_cvt(
    const float* __restrict__ W, __bf16* __restrict__ Wt, int R, int C) {
  int idx = blockIdx.x * 256 + threadIdx.x;
  int r8 = R >> 3;
  if (idx >= C * r8) return;
  int c = idx / r8;
  int r0 = (idx % r8) * 8;
  const float* src = W + (size_t)blockIdx.y * R * C;
  __bf16* dst = Wt + (size_t)blockIdx.y * R * C;
  bf16x8 v;
#pragma unroll
  for (int j = 0; j < 8; ++j) v[j] = (__bf16)src[(size_t)(r0 + j) * C + c];
  *(bf16x8*)(dst + (size_t)c * R + r0) = v;
}

// ---------------------------------------------------------------------------
// Kernel 1: per-head QKV projection from bf16 X and transposed bf16 weights.
// One wave: 16 rows (s) x 64 cols (full head). grid (64, 12, 3), block 128.
__global__ __launch_bounds__(128) void mha_qkv_proj(
    const __bf16* __restrict__ Xq, const __bf16* __restrict__ Xk, const __bf16* __restrict__ Xv,
    const __bf16* __restrict__ Wqt, const __bf16* __restrict__ Wkt, const __bf16* __restrict__ Wvt,
    const float* __restrict__ Bq, const float* __restrict__ Bk, const float* __restrict__ Bv,
    __bf16* __restrict__ qb, __bf16* __restrict__ kbuf, __bf16* __restrict__ vT) {
  const int lane = threadIdx.x & 31;
  const int wave = threadIdx.x >> 5;
  const int mtile = blockIdx.x * 4 + wave;  // 0..255 over B*S/16
  const int h = blockIdx.y;
  const int t = blockIdx.z;

  const __bf16* X  = (t == 0) ? Xq : (t == 1) ? Xk : Xv;
  const __bf16* Wt = (t == 0) ? Wqt : (t == 1) ? Wkt : Wvt;
  const float*  Bi = (t == 0) ? Bq : (t == 1) ? Bk : Bv;

  const int lr  = lane & 15;
  const int kga = (lane & 16) >> 1;  // A lane-group k offset (0/8)
  const int kgb = (lane & 16) >> 1;  // B lane-group k offset (0/8)
  const int m0 = mtile * 16;

  const __bf16* a_base = X + (size_t)(m0 + lr) * E_DIM + kga;
  // Wt[h][d][e]: per-lane contiguous e.
  const __bf16* w_base = Wt + (size_t)h * (E_DIM * D_DIM) + (size_t)lr * E_DIM + kgb;

  f32x8 acc[4];
#pragma unroll
  for (int j = 0; j < 4; ++j) acc[j] = fzero8();

  for (int kb = 0; kb < E_DIM; kb += 32) {
    // Hoist all loads of this K-step so they issue as one clause and overlap
    // with the WMMA group (partial s_wait_loadcnt instead of full waits).
    bf16x16 a = load_a_bf16(a_base + kb);
    bf16x16 bm[4];
#pragma unroll
    for (int j = 0; j < 4; ++j)
      bm[j] = load_b_bf16(w_base + (size_t)j * 16 * E_DIM + kb);
#pragma unroll
    for (int j = 0; j < 4; ++j) acc[j] = wmma_bf16f32(a, bm[j], acc[j]);
  }

  const int bb = m0 >> 11;       // batch
  const int s0 = m0 & 2047;      // row within batch
  const int mg = (lane & 16) >> 1;
  const size_t plane = ((size_t)bb * H_NUM + h) * PLANE;

  if (t == 2) {
    // v stored transposed: vT[b,h][d][s]; regs are consecutive in s -> 16B store.
#pragma unroll
    for (int j = 0; j < 4; ++j) {
      const float bias = Bi[h * D_DIM + j * 16 + lr];
      bf16x8 pk;
#pragma unroll
      for (int r = 0; r < 8; ++r) pk[r] = (__bf16)(acc[j][r] + bias);
      *(bf16x8*)(vT + plane + (size_t)(j * 16 + lr) * S_LEN + s0 + mg) = pk;
    }
  } else {
    __bf16* dst = ((t == 0) ? qb : kbuf) + plane;
    const float scale = (t == 0) ? 0.125f : 1.0f;  // fold 1/sqrt(D) into q
#pragma unroll
    for (int j = 0; j < 4; ++j) {
      const float bias = Bi[h * D_DIM + j * 16 + lr];
#pragma unroll
      for (int r = 0; r < 8; ++r)
        dst[(size_t)(s0 + r + mg) * D_DIM + j * 16 + lr] =
            (__bf16)((acc[j][r] + bias) * scale);
    }
  }
}

// ---------------------------------------------------------------------------
// Kernel 2: flash attention per (b,h). One wave owns 16 query rows, 32 keys
// per iteration. S^T = K*q^T so the probability fragment is already in the
// A-operand layout for P*V. V loads are issued before the softmax block so
// their latency hides under the exp/reduction VALU work.
// grid (32, 24), block 128.
__global__ __launch_bounds__(128) void mha_attention(
    const __bf16* __restrict__ qb, const __bf16* __restrict__ kbuf,
    const __bf16* __restrict__ vT, __bf16* __restrict__ o) {
  const int lane = threadIdx.x & 31;
  const int wave = threadIdx.x >> 5;
  const int qt = blockIdx.x * 4 + wave;  // 0..127 query tiles
  const int bh = blockIdx.y;             // b*H + h

  const size_t plane = (size_t)bh * PLANE;
  const __bf16* Q = qb + plane;
  const __bf16* K = kbuf + plane;
  const __bf16* V = vT + plane;  // [d][s]

  const int lr  = lane & 15;
  const int kga = (lane & 16) >> 1;
  const int kgb = (lane & 16) >> 1;
  const int mg  = (lane & 16) >> 1;

  // Preload q^T B-fragments (e in [0,32), [32,64)); scale already folded in.
  const __bf16* qbase = Q + (size_t)(qt * 16 + lr) * D_DIM + kgb;
  const bf16x16 qB0 = load_b_bf16(qbase);
  const bf16x16 qB1 = load_b_bf16(qbase + 32);

  f32x8 acc[4];
#pragma unroll
  for (int j = 0; j < 4; ++j) acc[j] = fzero8();
  float m_run = -1e30f, l_run = 0.0f;

  const __bf16* krow = K + (size_t)lr * D_DIM + kga;
  const __bf16* vrow = V + (size_t)lr * S_LEN + kgb;  // + j*16*S_LEN per n-tile

  for (int t = 0; t < S_LEN / 32; ++t) {
    const __bf16* kb0 = krow + (size_t)t * 32 * D_DIM;
    const __bf16* kb1 = kb0 + 16 * D_DIM;
    // Prefetch next iteration's K/V tiles into cache (global_prefetch_b8).
    __builtin_prefetch(kb0 + 32 * D_DIM, 0, 0);
    __builtin_prefetch(vrow + t * 32 + 32, 0, 0);

    // Hoist all K A-fragments, then the 4 score WMMAs.
    bf16x16 a00 = load_a_bf16(kb0);
    bf16x16 a01 = load_a_bf16(kb0 + 32);
    bf16x16 a10 = load_a_bf16(kb1);
    bf16x16 a11 = load_a_bf16(kb1 + 32);

    // Issue V B-fragment loads now: independent of scores and softmax, so
    // their latency overlaps the WMMAs below and the exp/reduction work.
    bf16x16 vB[4];
#pragma unroll
    for (int j = 0; j < 4; ++j)
      vB[j] = load_b_bf16(vrow + (size_t)j * 16 * S_LEN + t * 32);

    f32x8 st0 = wmma_bf16f32(a00, qB0, fzero8());
    f32x8 st1 = wmma_bf16f32(a10, qB0, fzero8());
    st0 = wmma_bf16f32(a01, qB1, st0);
    st1 = wmma_bf16f32(a11, qB1, st1);

    // Row (query) stats: lane holds keys {kg..kg+7, 16+kg..16+kg+7} of query
    // row (lane&15); partner lane is lane^16.
    float tm = st0[0];
#pragma unroll
    for (int r = 0; r < 8; ++r) { tm = fmaxf(tm, st0[r]); tm = fmaxf(tm, st1[r]); }
    tm = fmaxf(tm, __shfl_xor(tm, 16, 32));
    const float m_new = fmaxf(m_run, tm);

    float p0[8], p1[8], psum = 0.f;
#pragma unroll
    for (int r = 0; r < 8; ++r) {
      p0[r] = __expf(st0[r] - m_new);
      p1[r] = __expf(st1[r] - m_new);
      psum += p0[r] + p1[r];
    }
    psum += __shfl_xor(psum, 16, 32);
    const float corr = __expf(m_run - m_new);
    l_run = l_run * corr + psum;
    m_run = m_new;

    // Broadcast per-row rescale factor into the C layout (m = r + mg).
    float cf[8];
#pragma unroll
    for (int r = 0; r < 8; ++r) cf[r] = __shfl(corr, r + mg, 32);
#pragma unroll
    for (int j = 0; j < 4; ++j)
#pragma unroll
      for (int r = 0; r < 8; ++r) acc[j][r] *= cf[r];

    // P fragment is already in A layout: k = (i&7) + 16*(i>=8) + 8*(lane>=16).
    bf16x16 aP;
#pragma unroll
    for (int i = 0; i < 8; ++i) { aP[i] = (__bf16)p0[i]; aP[i + 8] = (__bf16)p1[i]; }

#pragma unroll
    for (int j = 0; j < 4; ++j) acc[j] = wmma_bf16f32(aP, vB[j], acc[j]);
  }

  // Normalize and store o[b,h,s,d] as bf16 (A-operand native for kernel 3).
  float lf[8];
#pragma unroll
  for (int r = 0; r < 8; ++r) lf[r] = 1.0f / __shfl(l_run, r + mg, 32);
  __bf16* O = o + plane + (size_t)qt * 16 * D_DIM;
#pragma unroll
  for (int j = 0; j < 4; ++j)
#pragma unroll
    for (int r = 0; r < 8; ++r)
      O[(size_t)(r + mg) * D_DIM + j * 16 + lr] = (__bf16)(acc[j][r] * lf[r]);
}

// ---------------------------------------------------------------------------
// Kernel 3: output projection. x = concat_heads(o) [B*S, E] @ Wo [E,E] + bo.
// One wave: 16 rows x 64 cols; head-aware A indexing (8-runs never cross a
// head boundary). grid (64, 12), block 128.
__global__ __launch_bounds__(128) void mha_out_proj(
    const __bf16* __restrict__ o, const __bf16* __restrict__ Wot,
    const float* __restrict__ bo, float* __restrict__ out) {
  const int lane = threadIdx.x & 31;
  const int wave = threadIdx.x >> 5;
  const int mtile = blockIdx.x * 4 + wave;  // 0..255
  const int nt = blockIdx.y;                // 0..11, 64 cols each

  const int lr  = lane & 15;
  const int kga = (lane & 16) >> 1;
  const int kgb = (lane & 16) >> 1;
  const int m0 = mtile * 16;
  const int bb = m0 >> 11;
  const int s0 = m0 & 2047;

  // Wot[n][e]: per-lane contiguous e.
  const __bf16* w_base = Wot + (size_t)(nt * 64 + lr) * E_DIM + kgb;

  f32x8 acc[4];
#pragma unroll
  for (int j = 0; j < 4; ++j) acc[j] = fzero8();

  for (int kb = 0; kb < E_DIM; kb += 32) {
    // A: x[m, e] = o[b, e/64, s, e%64]; each 8-run stays within one head.
    const int e0 = kb + kga;
    const int e1 = e0 + 16;
    const __bf16* ap0 =
        o + (((size_t)bb * H_NUM + (e0 >> 6)) * S_LEN + s0 + lr) * D_DIM + (e0 & 63);
    const __bf16* ap1 =
        o + (((size_t)bb * H_NUM + (e1 >> 6)) * S_LEN + s0 + lr) * D_DIM + (e1 & 63);
    bf16x8 lo = *(const bf16x8*)ap0;
    bf16x8 hi = *(const bf16x8*)ap1;
    bf16x16 a = __builtin_shufflevector(lo, hi, 0, 1, 2, 3, 4, 5, 6, 7,
                                        8, 9, 10, 11, 12, 13, 14, 15);
    bf16x16 bW[4];
#pragma unroll
    for (int j = 0; j < 4; ++j)
      bW[j] = load_b_bf16(w_base + (size_t)j * 16 * E_DIM + kb);
#pragma unroll
    for (int j = 0; j < 4; ++j) acc[j] = wmma_bf16f32(a, bW[j], acc[j]);
  }

  const int mg = (lane & 16) >> 1;
  float* Ob = out + ((size_t)bb * S_LEN + s0) * E_DIM + nt * 64;
#pragma unroll
  for (int j = 0; j < 4; ++j) {
    const float bias = bo[nt * 64 + j * 16 + lr];
#pragma unroll
    for (int r = 0; r < 8; ++r)
      Ob[(size_t)(r + mg) * E_DIM + j * 16 + lr] = acc[j][r] + bias;
  }
}

// ---------------------------------------------------------------------------
extern "C" void kernel_launch(void* const* d_in, const int* in_sizes, int n_in,
                              void* d_out, int out_size, void* d_ws, size_t ws_size,
                              hipStream_t stream) {
  const float* query = (const float*)d_in[0];
  const float* key_  = (const float*)d_in[1];
  const float* value = (const float*)d_in[2];
  const float* Wq = (const float*)d_in[3];
  const float* bq = (const float*)d_in[4];
  const float* Wk = (const float*)d_in[5];
  const float* bk = (const float*)d_in[6];
  const float* Wv = (const float*)d_in[7];
  const float* bv = (const float*)d_in[8];
  const float* Wo = (const float*)d_in[9];
  const float* bo = (const float*)d_in[10];

  // Workspace layout (bf16 elements); total ~48.8 MB.
  __bf16* xq  = (__bf16*)d_ws;
  __bf16* xk  = xq + X_SZ;
  __bf16* xv  = xk + X_SZ;
  __bf16* wqt = xv + X_SZ;
  __bf16* wkt = wqt + W_SZ;
  __bf16* wvt = wkt + W_SZ;
  __bf16* wot = wvt + W_SZ;
  __bf16* qb  = wot + WO_SZ;
  __bf16* kb  = qb + P_SZ;
  __bf16* vt  = kb + P_SZ;
  __bf16* ob  = vt + P_SZ;

  dim3 blk128(128), blk256(256);

  // Prep: activations -> bf16; weights -> transposed bf16.
  const int xn8 = (int)(X_SZ / 8);
  mha_cvt_bf16<<<dim3((xn8 + 255) / 256), blk256, 0, stream>>>(query, xq, xn8);
  mha_cvt_bf16<<<dim3((xn8 + 255) / 256), blk256, 0, stream>>>(key_, xk, xn8);
  mha_cvt_bf16<<<dim3((xn8 + 255) / 256), blk256, 0, stream>>>(value, xv, xn8);
  const int wthreads = D_DIM * (E_DIM / 8);  // per head matrix
  mha_transpose_cvt<<<dim3((wthreads + 255) / 256, H_NUM), blk256, 0, stream>>>(
      Wq, wqt, E_DIM, D_DIM);
  mha_transpose_cvt<<<dim3((wthreads + 255) / 256, H_NUM), blk256, 0, stream>>>(
      Wk, wkt, E_DIM, D_DIM);
  mha_transpose_cvt<<<dim3((wthreads + 255) / 256, H_NUM), blk256, 0, stream>>>(
      Wv, wvt, E_DIM, D_DIM);
  const int wothreads = E_DIM * (E_DIM / 8);
  mha_transpose_cvt<<<dim3((wothreads + 255) / 256, 1), blk256, 0, stream>>>(
      Wo, wot, E_DIM, E_DIM);

  // Main pipeline.
  mha_qkv_proj<<<dim3((B_NUM * S_LEN / 16) / 4, H_NUM, 3), blk128, 0, stream>>>(
      xq, xk, xv, wqt, wkt, wvt, bq, bk, bv, qb, kb, vt);
  mha_attention<<<dim3((S_LEN / 16) / 4, B_NUM * H_NUM), blk128, 0, stream>>>(
      qb, kb, vt, ob);
  mha_out_proj<<<dim3((B_NUM * S_LEN / 16) / 4, E_DIM / 64), blk128, 0, stream>>>(
      ob, wot, bo, (float*)d_out);
}
